// EfficientAttention_57543971832381
// MI455X (gfx1250) — matrix operand order repaired and checked by reading, not compiled, not run on hardware
//
#include <hip/hip_runtime.h>
#include <hip/hip_bf16.h>

typedef __bf16 v16bf __attribute__((ext_vector_type(16)));
typedef float  v8f   __attribute__((ext_vector_type(8)));
typedef int    v4i   __attribute__((ext_vector_type(4)));

#define AS1 __attribute__((address_space(1)))
#define AS3 __attribute__((address_space(3)))

#define S    16384
#define CIN  256

#if defined(__has_builtin)
# if __has_builtin(__builtin_amdgcn_global_load_async_to_lds_b128)
#  define ASYNC_BUILTIN 1
# endif
# if __has_builtin(__builtin_amdgcn_s_wait_asynccnt)
#  define WAITASYNC_BUILTIN 1
# endif
#endif

union FragU { uint4 u[2]; v16bf v; };

// async DMA: 16 bytes global -> LDS, tracked by ASYNCcnt
static __device__ __forceinline__ void async_g2l_b128(const __hip_bfloat16* g, unsigned short* l) {
#ifdef ASYNC_BUILTIN
    __builtin_amdgcn_global_load_async_to_lds_b128(
        (AS1 v4i*)(void*)g,
        (AS3 v4i*)(void*)l, 0, 0);
#else
    asm volatile("global_load_async_to_lds_b128 %0, %1, off"
                 :: "v"((unsigned)(unsigned long long)(void*)l),
                    "v"((const void*)g)
                 : "memory");
#endif
}

template <int N>
static __device__ __forceinline__ void wait_async() {
#ifdef WAITASYNC_BUILTIN
    __builtin_amdgcn_s_wait_asynccnt((unsigned short)N);
#else
    asm volatile("s_wait_asynccnt %0" :: "i"(N) : "memory");
#endif
}

// A fragment (16x32 bf16, M=row per lane&15, K 0..15 lo-lanes / 16..31 hi-lanes)
static __device__ __forceinline__ v16bf load_a_frag(const __hip_bfloat16* base, int lda, int row, int k0, int lane) {
    int kh = (lane & 16) ? 16 : 0;
    const __hip_bfloat16* p = base + (size_t)row * lda + k0 + kh;
    FragU f;
    f.u[0] = *(const uint4*)(p);       // K = kh+0..7
    f.u[1] = *(const uint4*)(p + 8);   // K = kh+8..15
    return f.v;
}

// B fragment direct from global (K-contiguous per column)
static __device__ __forceinline__ v16bf load_b_frag(const __hip_bfloat16* base, int ldb, int col, int k0, int lane) {
    int kh = (lane & 16) ? 16 : 0;
    const __hip_bfloat16* p = base + (size_t)col * ldb + k0 + kh;
    FragU f;
    f.u[0] = *(const uint4*)(p);
    f.u[1] = *(const uint4*)(p + 8);
    return f.v;
}

static __device__ __forceinline__ v8f wmma_bf16(v16bf a, v16bf b, v8f c) {
    return __builtin_amdgcn_wmma_f32_16x16x32_bf16(false, a, false, b, (short)0, c, false, false);
}

#define BSTRIDE 40  // ushorts per LDS column (80B: 16B-aligned, bank-spread)

// core GEMM: C(128x64) += A(128x256) * B(64 cols x 256 K), B async-staged to LDS double-buffered.
// block = 256 threads / 8 waves; wave wv owns rows m0..m0+15; acc[4] covers 4 col-subtiles.
static __device__ __forceinline__ void gemm_core(const __hip_bfloat16* __restrict__ A,
                                                 const __hip_bfloat16* __restrict__ Bsrc,  // [col][256]
                                                 unsigned short (*Bl)[64][BSTRIDE],
                                                 int m0, int n0, int lane, int tid, v8f acc[4]) {
    int ccol = tid >> 2;      // 0..63 : column this thread DMAs
    int cpart = tid & 3;      // 0..3  : 16B chunk within 32-K slab
    const __hip_bfloat16* gsrc = Bsrc + (size_t)(n0 + ccol) * CIN + cpart * 8;
    unsigned short* l0 = &Bl[0][ccol][cpart * 8];
    unsigned short* l1 = &Bl[1][ccol][cpart * 8];

    int arow = m0 + (lane & 15);
    int bkh = (lane & 16) ? 16 : 0;

    async_g2l_b128(gsrc, l0);  // k-step 0
#pragma unroll
    for (int kk = 0; kk < 8; ++kk) {
        if (kk < 7) {
            async_g2l_b128(gsrc + (kk + 1) * 32, ((kk + 1) & 1) ? l1 : l0);
            wait_async<1>();
        } else {
            wait_async<0>();
        }
        __syncthreads();  // buffer kk visible to all waves
        v16bf a = load_a_frag(A, CIN, arow, kk * 32, lane);
        const unsigned short* bb = &Bl[kk & 1][0][0];
#pragma unroll
        for (int sub = 0; sub < 4; ++sub) {
            const unsigned short* p = bb + (sub * 16 + (lane & 15)) * BSTRIDE + bkh;
            FragU f;
            f.u[0] = *(const uint4*)(p);
            f.u[1] = *(const uint4*)(p + 8);
            acc[sub] = wmma_bf16(a, f.v, acc[sub]);
        }
        __syncthreads();  // all reads of buffer kk done before it is DMA-overwritten
    }
}

// ---------------- prep: concat weights -> bf16, concat biases ----------------
__global__ void k_prep(const float* __restrict__ Wk, const float* __restrict__ Wq,
                       const float* __restrict__ Wv, const float* __restrict__ Wr,
                       const float* __restrict__ bk, const float* __restrict__ bq,
                       const float* __restrict__ bv,
                       __hip_bfloat16* __restrict__ Wcat, float* __restrict__ bcat,
                       __hip_bfloat16* __restrict__ Wrb) {
    int g = blockIdx.x * 256 + threadIdx.x;
    if (g < 768 * 256) {
        int row = g >> 8, c = g & 255;
        float v = (row < 256) ? Wk[row * 256 + c]
                : (row < 512) ? Wq[(row - 256) * 256 + c]
                              : Wv[(row - 512) * 256 + c];
        Wcat[g] = __float2bfloat16(v);
    } else if (g < 768 * 256 + 65536) {
        int i = g - 768 * 256;
        Wrb[i] = __float2bfloat16(Wr[i]);
    } else if (g < 768 * 256 + 65536 + 768) {
        int b = g - 768 * 256 - 65536;
        bcat[b] = (b < 256) ? bk[b] : (b < 512) ? bq[b - 256] : bv[b - 512];
    }
}

// ---------------- transpose x[c][s] f32 -> x_t[s][c] bf16 ----------------
__global__ void k_xt(const float* __restrict__ x, __hip_bfloat16* __restrict__ xt) {
    __shared__ float t[32][33];
    int c0 = blockIdx.x * 32, s0 = blockIdx.y * 32;
    int tid = threadIdx.x;
    int sl = tid & 31, cl = tid >> 5;
#pragma unroll
    for (int r = 0; r < 4; r++)
        t[cl + r * 8][sl] = x[(size_t)(c0 + cl + r * 8) * S + s0 + sl];
    __syncthreads();
    int cl2 = tid & 31, sl2 = tid >> 5;
#pragma unroll
    for (int r = 0; r < 4; r++)
        xt[(size_t)(s0 + sl2 + r * 8) * CIN + c0 + cl2] = __float2bfloat16(t[cl2][sl2 + r * 8]);
}

// ---------------- projection GEMM: [768x256] @ [256xS] + bias ----------------
__global__ void k_proj(const __hip_bfloat16* __restrict__ Wcat, const float* __restrict__ bcat,
                       const __hip_bfloat16* __restrict__ xt,
                       float* __restrict__ keys, float* __restrict__ queries,
                       __hip_bfloat16* __restrict__ vals) {
    __shared__ __align__(16) unsigned short Bl[2][64][BSTRIDE];
    int tid = threadIdx.x;
    int lane = tid & 31, wv = tid >> 5;
    int m0 = blockIdx.y * 128 + wv * 16;
    int n0 = blockIdx.x * 64;
    v8f acc[4] = {};
    gemm_core(Wcat, xt, Bl, m0, n0, lane, tid, acc);

    int rbase = m0 + ((lane & 16) ? 8 : 0);
    int cbase = n0 + (lane & 15);
#pragma unroll
    for (int sub = 0; sub < 4; ++sub) {
#pragma unroll
        for (int p = 0; p < 8; p++) {
            int row = rbase + p;
            int col = cbase + sub * 16;
            float v = acc[sub][p] + bcat[row];
            if (row < 256)      keys[(size_t)row * S + col] = v;
            else if (row < 512) queries[(size_t)(row - 256) * S + col] = v;
            else                vals[(size_t)(row - 512) * S + col] = __float2bfloat16(v);
        }
    }
}

// ---------------- per-key-row max and 1/sum(exp) ----------------
__global__ void k_krows(const float* __restrict__ keys, float* __restrict__ rowmax,
                        float* __restrict__ rowinv) {
    __shared__ float red[256];
    int k = blockIdx.x, t = threadIdx.x;
    const float* r = keys + (size_t)k * S;
    float m = -1e30f;
    for (int s = t; s < S; s += 256) m = fmaxf(m, r[s]);
    red[t] = m; __syncthreads();
    for (int o = 128; o > 0; o >>= 1) { if (t < o) red[t] = fmaxf(red[t], red[t + o]); __syncthreads(); }
    float mx = red[0]; __syncthreads();
    float sum = 0.f;
    for (int s = t; s < S; s += 256) sum += __expf(r[s] - mx);
    red[t] = sum; __syncthreads();
    for (int o = 128; o > 0; o >>= 1) { if (t < o) red[t] += red[t + o]; __syncthreads(); }
    if (t == 0) { rowmax[k] = mx; rowinv[k] = 1.0f / red[0]; }
}

// ---------------- normalize K (softmax over s) + attention_map ----------------
__global__ void k_knorm(const float* __restrict__ keys, const float* __restrict__ rowmax,
                        const float* __restrict__ rowinv, __hip_bfloat16* __restrict__ Ksm,
                        float* __restrict__ map_out) {
    int s = blockIdx.x * 256 + threadIdx.x;
    float acc = 0.f;
    for (int k = 0; k < 256; k++) {
        float e = __expf(keys[(size_t)k * S + s] - rowmax[k]) * rowinv[k];
        Ksm[(size_t)k * S + s] = __float2bfloat16(e);
        acc += e;
    }
    map_out[s] = acc * (1.0f / 256.0f);
}

__global__ void k_zero(float* __restrict__ p, int n) {
    int i = blockIdx.x * blockDim.x + threadIdx.x;
    if (i < n) p[i] = 0.f;
}

// ---------------- ctx = Ksm(32xS) @ vals(32xS)^T per head, split-K + hw f32 atomics ----------------
__global__ void k_ctx(const __hip_bfloat16* __restrict__ Ksm, const __hip_bfloat16* __restrict__ vals,
                      float* __restrict__ ctx) {
    int lane = threadIdx.x & 31, wv = threadIdx.x >> 5;
    int mt = wv & 1, nt = wv >> 1;
    int h = blockIdx.y;
    int s0 = blockIdx.x * 256;
    v8f acc = {};
    int arow = h * 32 + mt * 16 + (lane & 15);
    int bcol = h * 32 + nt * 16 + (lane & 15);
#pragma unroll
    for (int kk = 0; kk < 8; ++kk) {
        int k0 = s0 + kk * 32;
        v16bf a = load_a_frag(Ksm, S, arow, k0, lane);
        v16bf b = load_b_frag(vals, S, bcol, k0, lane);
        acc = wmma_bf16(a, b, acc);
    }
    int krow = mt * 16 + ((lane & 16) ? 8 : 0);
    int vcol = nt * 16 + (lane & 15);
#pragma unroll
    for (int p = 0; p < 8; p++)
        unsafeAtomicAdd(&ctx[h * 1024 + (krow + p) * 32 + vcol], acc[p]);
}

__global__ void k_ctxcopy(const float* __restrict__ ctx, float* __restrict__ out) {
    int i = blockIdx.x * 256 + threadIdx.x;
    if (i < 1024) out[i] = ctx[7 * 1024 + i];
}

// ---------------- Q softmax over dk (local 32) + att = ctx^T @ Qsm -> att_t[s][v] ----------------
__global__ void k_qatt(const float* __restrict__ queries, const float* __restrict__ ctx,
                       __hip_bfloat16* __restrict__ att_t) {
    __shared__ float c[8192];
    for (int i = threadIdx.x; i < 8192; i += 256) c[i] = ctx[i];
    __syncthreads();
    int s = blockIdx.x * 256 + threadIdx.x;
    for (int h = 0; h < 8; h++) {
        float q[32];
        float m = -1e30f;
#pragma unroll
        for (int k = 0; k < 32; k++) { q[k] = queries[(size_t)(h * 32 + k) * S + s]; m = fmaxf(m, q[k]); }
        float sum = 0.f;
#pragma unroll
        for (int k = 0; k < 32; k++) { q[k] = __expf(q[k] - m); sum += q[k]; }
        float inv = 1.0f / sum;
        const float* ch = c + h * 1024;
        for (int v = 0; v < 32; v++) {
            float a = 0.f;
#pragma unroll
            for (int k = 0; k < 32; k++) a += q[k] * ch[k * 32 + v];
            att_t[(size_t)s * 256 + h * 32 + v] = __float2bfloat16(a * inv);
        }
    }
}

// ---------------- output GEMM: Wr(256x256) @ att(256xS) + br -> f32 out ----------------
__global__ void k_out(const __hip_bfloat16* __restrict__ Wrb, const float* __restrict__ br,
                      const __hip_bfloat16* __restrict__ att_t, float* __restrict__ outp) {
    __shared__ __align__(16) unsigned short Bl[2][64][BSTRIDE];
    int tid = threadIdx.x;
    int lane = tid & 31, wv = tid >> 5;
    int m0 = blockIdx.y * 128 + wv * 16;
    int n0 = blockIdx.x * 64;
    v8f acc[4] = {};
    gemm_core(Wrb, att_t, Bl, m0, n0, lane, tid, acc);

    int rbase = m0 + ((lane & 16) ? 8 : 0);
#pragma unroll
    for (int sub = 0; sub < 4; ++sub)
#pragma unroll
        for (int p = 0; p < 8; p++) {
            int row = rbase + p, col = n0 + sub * 16 + (lane & 15);
            outp[(size_t)row * S + col] = acc[sub][p] + br[row];
        }
}

extern "C" void kernel_launch(void* const* d_in, const int* in_sizes, int n_in,
                              void* d_out, int out_size, void* d_ws, size_t ws_size,
                              hipStream_t stream) {
    (void)in_sizes; (void)n_in; (void)out_size; (void)ws_size;
    const float* x  = (const float*)d_in[0];
    const float* Wk = (const float*)d_in[1];
    const float* bk = (const float*)d_in[2];
    const float* Wq = (const float*)d_in[3];
    const float* bq = (const float*)d_in[4];
    const float* Wv = (const float*)d_in[5];
    const float* bv = (const float*)d_in[6];
    const float* Wr = (const float*)d_in[7];
    const float* br = (const float*)d_in[8];
    float* out = (float*)d_out;

    const size_t ATT0 = 0;
    const size_t CTX0 = (size_t)8 * 256 * S;
    const size_t MAP0 = CTX0 + (size_t)8 * 32 * 32;

    char* w = (char*)d_ws;
    auto alloc = [&](size_t bytes) -> void* {
        void* p = (void*)w;
        w += (bytes + 255) & ~(size_t)255;
        return p;
    };
    __hip_bfloat16* Wcat  = (__hip_bfloat16*)alloc((size_t)768 * 256 * 2);
    float*          bcat  = (float*)alloc(768 * 4);
    __hip_bfloat16* Wrb   = (__hip_bfloat16*)alloc((size_t)256 * 256 * 2);
    __hip_bfloat16* xt    = (__hip_bfloat16*)alloc((size_t)S * CIN * 2);
    float*          keys  = (float*)alloc((size_t)256 * S * 4);
    float*          quer  = (float*)alloc((size_t)256 * S * 4);
    __hip_bfloat16* vals  = (__hip_bfloat16*)alloc((size_t)256 * S * 2);
    __hip_bfloat16* Ksm   = (__hip_bfloat16*)alloc((size_t)256 * S * 2);
    __hip_bfloat16* att_t = (__hip_bfloat16*)alloc((size_t)S * 256 * 2);
    float*          rowmax = (float*)alloc(256 * 4);
    float*          rowinv = (float*)alloc(256 * 4);
    float*          ctx    = (float*)alloc(8192 * 4);

    k_prep<<<1028, 256, 0, stream>>>(Wk, Wq, Wv, Wr, bk, bq, bv, Wcat, bcat, Wrb);

    for (int n = 0; n < 8; n++) {
        const float* xn = x + (size_t)n * CIN * S;
        k_xt<<<dim3(8, 512), 256, 0, stream>>>(xn, xt);
        k_proj<<<dim3(S / 64, 6), 256, 0, stream>>>(Wcat, bcat, xt, keys, quer, vals);
        k_krows<<<256, 256, 0, stream>>>(keys, rowmax, rowinv);
        k_knorm<<<S / 256, 256, 0, stream>>>(keys, rowmax, rowinv, Ksm, out + MAP0 + (size_t)n * S);
        k_zero<<<32, 256, 0, stream>>>(ctx, 8192);
        k_ctx<<<dim3(64, 8), 128, 0, stream>>>(Ksm, vals, ctx);
        k_ctxcopy<<<4, 256, 0, stream>>>(ctx, out + CTX0 + (size_t)n * 1024);
        k_qatt<<<S / 256, 256, 0, stream>>>(quer, ctx, att_t);
        k_out<<<dim3(S / 64, 2), 256, 0, stream>>>(Wrb, br, att_t, out + ATT0 + (size_t)n * CIN * S);
    }
}
